// ViT_32452772889258
// MI455X (gfx1250) — matrix operand order repaired and checked by reading, not compiled, not run on hardware
//
// MI455X (gfx1250) transformer encoder-decoder forward pass.
//
// Compute-bound (~2 TFLOP of GEMM vs ~sub-GB of traffic at 23.3 TB/s), so all
// dense math is routed through v_wmma_f32_16x16x32_bf16 (f32 inputs converted
// to bf16 in LDS, f32 accumulation). One generic batched/strided WMMA GEMM
// kernel with a fused epilogue (bias / residual-accumulate / BatchNorm / ReLU)
// implements: conv1d(k=5) as 5 shifted accumulated GEMMs, all projections,
// FFNs, attention QK^T and AV (batched over B*H), and the final head.
// Softmax + LayerNorm are wave32 shuffle-reduction kernels.
//
// Workspace requirement: 7 * 32MB activation buffers + 128MB FF/scores buffer
// = ~369MB of d_ws.
//
// WMMA operand layouts per CDNA5 ISA 7.12.2 (wave32):
//   A 16x32 bf16: lanes 0-15 row=lane, elems 0-7 -> K0-7, 8-15 -> K16-23;
//                 lanes 16-31 row=lane-16, elems -> K8-15, K24-31.
//   B 32x16 bf16: lane = N col (mod 16); K split as A: lanes 0-15 K0-7, K16-23;
//                 lanes 16-31 -> K8-15, K24-31.
//   C/D 16x16 f32: VGPR r, lane l -> M = r + 8*(l>=16), N = l&15.
// LDS tiles are stored [row][K] (A) and [col][K] (B-transposed) with rows
// padded to 40 bf16 (80B = 5 x 16B) so fragments are two aligned b128 DS loads
// and the 16-lane row fan-out hits distinct banks.
//
// Staging: per-thread row-remap/head-term address math is hoisted out of the
// K loop; when the K-walk is unit-stride and fully in bounds (A almost always;
// B in the QK^T GEMMs) the 16-element run is fetched as 4x global_load_b128.

#include <hip/hip_runtime.h>
#include <cstring>

typedef __attribute__((ext_vector_type(16))) __bf16 v16bf;
typedef __attribute__((ext_vector_type(8)))  float  v8f;

#define LDSW 40   // bf16 elements per padded LDS row (32 data + 8 pad)

struct GemmP {
  const float* A; const float* B; const float* bias; const float* residual;
  float* C;
  int M, N, K;
  long long lda, ka;            // A elem (r,k) -> rowmap(r)*lda + k*ka
  int tOut, tIn, shift;         // rowmap: tOut==0 -> r ; else (r/tOut)*tIn + r%tOut + shift
  long long kSB, nSB, headN, hStr; // B elem (k,n) -> k*kSB + (n/headN)*hStr + (n%headN)*nSB
  long long ldc;
  int Hdiv;                     // grid.z -> b = z/Hdiv, h = z%Hdiv
  long long aOffB, aOffH, bOffB, bOffH, cOffB, cOffH;
  int relu;
  const float* bn_rm; const float* bn_rv; const float* bn_w; const float* bn_b;
};

__device__ __forceinline__ unsigned short f2bf(float f) {
  unsigned int u = __float_as_uint(f);
  u += 0x7FFFu + ((u >> 16) & 1u);      // round-to-nearest-even
  return (unsigned short)(u >> 16);
}

union Frag { uint4 u[2]; v16bf v; };

__global__ void __launch_bounds__(256) gemm_bf16_wmma(GemmP p) {
  __shared__ __align__(16) unsigned short sA[128 * LDSW];
  __shared__ __align__(16) unsigned short sB[128 * LDSW];

  const int tid  = threadIdx.x;
  const int lane = tid & 31;
  const int wave = tid >> 5;
  const int wm   = wave >> 1;      // 4 waves along M (32 rows each)
  const int wn   = wave & 1;       // 2 waves along N (64 cols each)
  const int rsel = lane & 15;
  const int hsel = lane >> 4;

  const int z = (int)blockIdx.z;
  const long long bb = z / p.Hdiv;
  const long long hh = z % p.Hdiv;
  const float* Ab = p.A + bb * p.aOffB + hh * p.aOffH;
  const float* Bb = p.B + bb * p.bOffB + hh * p.bOffH;
  float*       Cb = p.C + bb * p.cOffB + hh * p.cOffH;
  const float* Rb = p.residual ? (p.residual + bb * p.cOffB + hh * p.cOffH)
                               : (const float*)0;

  const int m0 = blockIdx.y * 128;
  const int n0 = blockIdx.x * 128;

  // ---- K-invariant staging coordinates (hoisted) ----
  const int sr  = tid >> 1;            // staged row (A) / col (B) 0..127
  const int skk = (tid & 1) * 16;      // K sub-offset 0 or 16
  const int gm  = m0 + sr;
  const bool mOk = (gm < p.M);
  long long aRow = 0;
  if (mOk)
    aRow = p.tOut ? ((long long)(gm / p.tOut) * p.tIn + (gm % p.tOut) + p.shift)
                  : (long long)gm;
  const float* aBase = Ab + aRow * p.lda;        // + k*ka
  unsigned short* aDst = sA + sr * LDSW + skk;

  const int gn = n0 + sr;
  const bool nOk = (gn < p.N);
  long long nTerm = 0;
  if (nOk)
    nTerm = (long long)(gn / p.headN) * p.hStr + (long long)(gn % p.headN) * p.nSB;
  const float* bBase = Bb + nTerm;               // + k*kSB
  unsigned short* bDst = sB + sr * LDSW + skk;

  const bool aUnit = (p.ka == 1);
  const bool bUnit = (p.kSB == 1);

  v8f acc[2][4];
  const v8f vzero = {0.f,0.f,0.f,0.f,0.f,0.f,0.f,0.f};
  for (int i = 0; i < 2; ++i)
    for (int j = 0; j < 4; ++j) acc[i][j] = vzero;

  for (int k0 = 0; k0 < p.K; k0 += 32) {
    const int kb = k0 + skk;
    // ---- stage A tile (128 x 32) as bf16, zero-padded OOB ----
    if (mOk && aUnit && (kb + 16 <= p.K)) {
      if (k0 + 32 < p.K) __builtin_prefetch(aBase + kb + 32, 0, 0);
      const float4* s = (const float4*)(aBase + kb);   // 16B-aligned by layout
      const float4 f0 = s[0], f1 = s[1], f2 = s[2], f3 = s[3];
      aDst[0]  = f2bf(f0.x); aDst[1]  = f2bf(f0.y); aDst[2]  = f2bf(f0.z); aDst[3]  = f2bf(f0.w);
      aDst[4]  = f2bf(f1.x); aDst[5]  = f2bf(f1.y); aDst[6]  = f2bf(f1.z); aDst[7]  = f2bf(f1.w);
      aDst[8]  = f2bf(f2.x); aDst[9]  = f2bf(f2.y); aDst[10] = f2bf(f2.z); aDst[11] = f2bf(f2.w);
      aDst[12] = f2bf(f3.x); aDst[13] = f2bf(f3.y); aDst[14] = f2bf(f3.z); aDst[15] = f2bf(f3.w);
    } else {
      for (int j = 0; j < 16; ++j) {
        const int gk = kb + j;
        float v = 0.f;
        if (mOk && gk < p.K) v = aBase[(long long)gk * p.ka];
        aDst[j] = f2bf(v);
      }
    }
    // ---- stage B tile (32 x 128), transposed into sB[col][k] ----
    if (nOk && bUnit && (kb + 16 <= p.K)) {
      const float4* s = (const float4*)(bBase + kb);
      const float4 f0 = s[0], f1 = s[1], f2 = s[2], f3 = s[3];
      bDst[0]  = f2bf(f0.x); bDst[1]  = f2bf(f0.y); bDst[2]  = f2bf(f0.z); bDst[3]  = f2bf(f0.w);
      bDst[4]  = f2bf(f1.x); bDst[5]  = f2bf(f1.y); bDst[6]  = f2bf(f1.z); bDst[7]  = f2bf(f1.w);
      bDst[8]  = f2bf(f2.x); bDst[9]  = f2bf(f2.y); bDst[10] = f2bf(f2.z); bDst[11] = f2bf(f2.w);
      bDst[12] = f2bf(f3.x); bDst[13] = f2bf(f3.y); bDst[14] = f2bf(f3.z); bDst[15] = f2bf(f3.w);
    } else {
      for (int j = 0; j < 16; ++j) {
        const int gk = kb + j;
        float v = 0.f;
        if (nOk && gk < p.K) v = bBase[(long long)gk * p.kSB];
        bDst[j] = f2bf(v);
      }
    }
    __syncthreads();

    // ---- fragments (ISA lane layouts) + 2x4 WMMA ----
    Frag af[2], bf[4];
    for (int tm = 0; tm < 2; ++tm) {
      const uint4* pa = (const uint4*)(sA + (wm * 32 + tm * 16 + rsel) * LDSW);
      af[tm].u[0] = pa[hsel];          // K 0-7   | 8-15
      af[tm].u[1] = pa[2 + hsel];      // K 16-23 | 24-31
    }
    for (int tn = 0; tn < 4; ++tn) {
      const uint4* pb = (const uint4*)(sB + (wn * 64 + tn * 16 + rsel) * LDSW);
      bf[tn].u[0] = pb[hsel];          // K 0-7   | 8-15
      bf[tn].u[1] = pb[2 + hsel];      // K 16-23 | 24-31
    }
    for (int tm = 0; tm < 2; ++tm)
      for (int tn = 0; tn < 4; ++tn)
        acc[tm][tn] = __builtin_amdgcn_wmma_f32_16x16x32_bf16(
            false, af[tm].v, false, bf[tn].v, (short)0, acc[tm][tn], false, false);
    __syncthreads();
  }

  // ---- epilogue: bias -> residual -> batchnorm -> relu -> store ----
  for (int tm = 0; tm < 2; ++tm)
    for (int tn = 0; tn < 4; ++tn)
      for (int r = 0; r < 8; ++r) {
        const int m  = m0 + wm * 32 + tm * 16 + r + hsel * 8;
        const int cn = n0 + wn * 64 + tn * 16 + rsel;
        if (m < p.M && cn < p.N) {
          float v = acc[tm][tn][r];
          if (p.bias) v += p.bias[cn];
          const long long ci = (long long)m * p.ldc + cn;
          if (Rb) v += Rb[ci];
          if (p.bn_w) {
            const float s = p.bn_w[cn] * rsqrtf(p.bn_rv[cn] + 1e-5f);
            v = (v - p.bn_rm[cn]) * s + p.bn_b[cn];
          }
          if (p.relu) v = fmaxf(v, 0.f);
          Cb[ci] = v;
        }
      }
}

// Row softmax with optional causal mask; writes zeros in masked region so the
// following AV GEMM can read a dense K range. One wave per row.
__global__ void __launch_bounds__(256) softmax_rows(float* S, int nRows, int T,
                                                    int Svalid, int ld,
                                                    int causal, float scale) {
  const int row  = blockIdx.x * 8 + (threadIdx.x >> 5);
  const int lane = threadIdx.x & 31;
  if (row >= nRows) return;
  const int t     = row % T;
  const int valid = causal ? (t + 1) : Svalid;
  float* p = S + (long long)row * ld;

  float mx = -3.4e38f;
  for (int s = lane; s < valid; s += 32) mx = fmaxf(mx, p[s] * scale);
  for (int o = 16; o > 0; o >>= 1) mx = fmaxf(mx, __shfl_xor(mx, o, 32));

  float sum = 0.f;
  for (int s = lane; s < valid; s += 32) {
    const float e = __expf(p[s] * scale - mx);
    p[s] = e;
    sum += e;
  }
  for (int o = 16; o > 0; o >>= 1) sum += __shfl_xor(sum, o, 32);
  const float inv = 1.0f / sum;
  for (int s = lane; s < valid; s += 32) p[s] *= inv;
  for (int s = valid + lane; s < Svalid; s += 32) p[s] = 0.f;
}

// LayerNorm over C=512, one wave per row (16 elems/lane). Safe in-place.
__global__ void __launch_bounds__(256) layernorm_rows(const float* X, float* Y,
                                                      const float* w, const float* b,
                                                      int nRows) {
  const int row  = blockIdx.x * 8 + (threadIdx.x >> 5);
  const int lane = threadIdx.x & 31;
  if (row >= nRows) return;
  const float* x = X + (long long)row * 512;
  float v[16];
  float sum = 0.f;
  for (int i = 0; i < 16; ++i) { v[i] = x[lane + i * 32]; sum += v[i]; }
  for (int o = 16; o > 0; o >>= 1) sum += __shfl_xor(sum, o, 32);
  const float mean = sum * (1.0f / 512.0f);
  float var = 0.f;
  for (int i = 0; i < 16; ++i) { const float d = v[i] - mean; var += d * d; }
  for (int o = 16; o > 0; o >>= 1) var += __shfl_xor(var, o, 32);
  const float inv = rsqrtf(var * (1.0f / 512.0f) + 1e-5f);
  float* y = Y + (long long)row * 512;
  for (int i = 0; i < 16; ++i) {
    const int c = lane + i * 32;
    y[c] = (v[i] - mean) * inv * w[c] + b[c];
  }
}

// Embedding gather: out[(b,t),c] = embed[idx[b,t]][c], float4 per thread.
__global__ void __launch_bounds__(256) embed_gather(const int* idx, const float* embed,
                                                    float* out, int nTok) {
  const int i   = blockIdx.x * 256 + threadIdx.x;
  const int tok = i >> 7;            // 128 float4 per 512-elem row
  const int c4  = i & 127;
  if (tok >= nTok) return;
  ((float4*)out)[i] =
      ((const float4*)(embed + (long long)idx[tok] * 512))[c4];
}

static inline GemmP gdef() {
  GemmP p;
  memset(&p, 0, sizeof(p));
  p.ka = 1; p.nSB = 1; p.headN = (1LL << 40); p.Hdiv = 1;
  return p;
}

static inline void launch_gemm(const GemmP& p, int batches, hipStream_t s) {
  dim3 grid((p.N + 127) / 128, (p.M + 127) / 128, batches);
  gemm_bf16_wmma<<<grid, dim3(256), 0, s>>>(p);
}

extern "C" void kernel_launch(void* const* d_in, const int* in_sizes, int n_in,
                              void* d_out, int out_size, void* d_ws, size_t ws_size,
                              hipStream_t stream) {
  (void)in_sizes; (void)n_in; (void)out_size; (void)ws_size;

  const int*   enc_idx = (const int*)  d_in[0];
  const float* dec_idx = (const float*)d_in[1];
  const float* embed   = (const float*)d_in[2];
  const float* conv_w  = (const float*)d_in[3];
  const float* conv_b  = (const float*)d_in[4];
  const float* bn_w    = (const float*)d_in[5];
  const float* bn_b    = (const float*)d_in[6];
  const float* bn_rm   = (const float*)d_in[7];
  const float* bn_rv   = (const float*)d_in[8];
  const float* epre_pw = (const float*)d_in[9];
  const float* epre_pb = (const float*)d_in[10];
  const float* dpre_w1 = (const float*)d_in[11];
  const float* dpre_b1 = (const float*)d_in[12];
  const float* dpre_w2 = (const float*)d_in[13];
  const float* dpre_b2 = (const float*)d_in[14];
  const float* dpre_pw = (const float*)d_in[15];
  const float* dpre_pb = (const float*)d_in[16];
  const float* e_ln1w  = (const float*)d_in[17];
  const float* e_ln1b  = (const float*)d_in[18];
  const float* e_qw    = (const float*)d_in[19];
  const float* e_kw    = (const float*)d_in[20];
  const float* e_vw    = (const float*)d_in[21];
  const float* e_pw    = (const float*)d_in[22];
  const float* e_pb    = (const float*)d_in[23];
  const float* e_ln2w  = (const float*)d_in[24];
  const float* e_ln2b  = (const float*)d_in[25];
  const float* e_f1w   = (const float*)d_in[26];
  const float* e_f1b   = (const float*)d_in[27];
  const float* e_f2w   = (const float*)d_in[28];
  const float* e_f2b   = (const float*)d_in[29];
  const float* d_ln1w  = (const float*)d_in[30];
  const float* d_ln1b  = (const float*)d_in[31];
  const float* d_sqw   = (const float*)d_in[32];
  const float* d_skw   = (const float*)d_in[33];
  const float* d_svw   = (const float*)d_in[34];
  const float* d_spw   = (const float*)d_in[35];
  const float* d_spb   = (const float*)d_in[36];
  const float* d_ln2w  = (const float*)d_in[37];
  const float* d_ln2b  = (const float*)d_in[38];
  const float* d_cqw   = (const float*)d_in[39];
  const float* d_ckw   = (const float*)d_in[40];
  const float* d_cvw   = (const float*)d_in[41];
  const float* d_cpw   = (const float*)d_in[42];
  const float* d_cpb   = (const float*)d_in[43];
  const float* d_ln3w  = (const float*)d_in[44];
  const float* d_ln3b  = (const float*)d_in[45];
  const float* d_f1w   = (const float*)d_in[46];
  const float* d_f1b   = (const float*)d_in[47];
  const float* d_f2w   = (const float*)d_in[48];
  const float* d_f2b   = (const float*)d_in[49];
  const float* m_w1    = (const float*)d_in[50];
  const float* m_b1    = (const float*)d_in[51];
  const float* m_w2    = (const float*)d_in[52];
  const float* m_b2    = (const float*)d_in[53];

  // ---- workspace layout (floats) ----
  const long long NB = 8388608LL;       // 16384 rows x 512
  float* ws   = (float*)d_ws;
  float* bX   = ws;                     // encoder x -> memory
  float* bH   = ws + 1 * NB;            // LN output / temporaries
  float* bQ   = ws + 2 * NB;
  float* bK   = ws + 3 * NB;
  float* bV   = ws + 4 * NB;
  float* bO   = ws + 5 * NB;            // attention output (pre-projection)
  float* bD   = ws + 6 * NB;            // decoder x
  float* bBIG = ws + 7 * NB;            // 33554432 floats: FF hidden / scores

  const int Te = 244;                   // encoder T after 3 VALID k=5 convs
  const int Td = 256;                   // decoder T
  const int Me = 64 * Te;               // 15616
  const int Md = 64 * Td;               // 16384
  const float SCALE = 0.04419417382415922f;   // 1/sqrt(512)

  auto linear = [&](const float* A, long long lda, const float* B, long long kSB,
                    const float* bias, const float* res, float* C, long long ldc,
                    int M, int N, int K, int relu) {
    GemmP p = gdef();
    p.A = A; p.lda = lda; p.B = B; p.kSB = kSB;
    p.bias = bias; p.residual = res;
    p.C = C; p.ldc = ldc; p.M = M; p.N = N; p.K = K; p.relu = relu;
    launch_gemm(p, 1, stream);
  };
  // headed projection: W[h][c][d], out col n = h*128+d; no bias.
  auto headlin = [&](const float* A, const float* W, float* C, int M) {
    GemmP p = gdef();
    p.A = A; p.lda = 512; p.B = W;
    p.kSB = 128; p.headN = 128; p.hStr = 512LL * 128; p.nSB = 1;
    p.C = C; p.ldc = 512; p.M = M; p.N = 512; p.K = 512;
    launch_gemm(p, 1, stream);
  };
  auto ln = [&](const float* X, float* Y, const float* w, const float* b, int rows) {
    layernorm_rows<<<dim3((rows + 7) / 8), dim3(256), 0, stream>>>(X, Y, w, b, rows);
  };
  // batched attention over (b,h): scores = Q K^T -> softmax -> O = P V
  auto attn = [&](const float* Qb, int Tq, const float* Kb, const float* Vb, int Tk,
                  float* Ob, int causal) {
    GemmP p = gdef();                      // scores (Tq x Tk), ld 256
    p.A = Qb; p.lda = 512; p.aOffB = (long long)Tq * 512; p.aOffH = 128;
    p.B = Kb; p.kSB = 1; p.nSB = 512; p.bOffB = (long long)Tk * 512; p.bOffH = 128;
    p.C = bBIG; p.ldc = 256; p.cOffH = (long long)Tq * 256; p.cOffB = 4LL * Tq * 256;
    p.M = Tq; p.N = Tk; p.K = 128; p.Hdiv = 4;
    launch_gemm(p, 64 * 4, stream);

    const int rows = 64 * 4 * Tq;
    softmax_rows<<<dim3((rows + 7) / 8), dim3(256), 0, stream>>>(
        bBIG, rows, Tq, Tk, 256, causal, SCALE);

    GemmP q = gdef();                      // O = P V (Tq x 128 per head)
    q.A = bBIG; q.lda = 256; q.aOffB = 4LL * Tq * 256; q.aOffH = (long long)Tq * 256;
    q.B = Vb; q.kSB = 512; q.nSB = 1; q.bOffB = (long long)Tk * 512; q.bOffH = 128;
    q.C = Ob; q.ldc = 512; q.cOffB = (long long)Tq * 512; q.cOffH = 128;
    q.M = Tq; q.N = 128; q.K = Tk; q.Hdiv = 4;
    launch_gemm(q, 64 * 4, stream);
  };
  // conv1d(k=5, VALID) as 5 shifted accumulated GEMMs; optional fused BN+ReLU.
  auto conv = [&](const float* in, float* out, int Tout, int Tin, int layer, int bnIdx) {
    for (int kk = 0; kk < 5; ++kk) {
      GemmP p = gdef();
      p.A = in; p.lda = 512;
      p.tOut = Tout; p.tIn = Tin; p.shift = kk;
      p.B = conv_w + (long long)layer * 512 * 512 * 5 + kk;  // w[co][ci][kk]
      p.kSB = 5; p.nSB = 2560;
      p.C = out; p.ldc = 512;
      p.M = 64 * Tout; p.N = 512; p.K = 512;
      if (kk == 0) p.bias = conv_b + layer * 512;
      else         p.residual = out;
      if (kk == 4 && bnIdx >= 0) {
        p.bn_rm = bn_rm + bnIdx * 512; p.bn_rv = bn_rv + bnIdx * 512;
        p.bn_w  = bn_w  + bnIdx * 512; p.bn_b  = bn_b  + bnIdx * 512;
        p.relu = 1;
      }
      launch_gemm(p, 1, stream);
    }
  };

  // ===== encoder frontend =====
  embed_gather<<<dim3(8192), dim3(256), 0, stream>>>(enc_idx, embed, bQ, 64 * 256);
  conv(bQ, bK, 252, 256, 0, -1);
  conv(bK, bV, 248, 252, 1, 0);
  conv(bV, bO, 244, 248, 2, 1);
  linear(bO, 512, epre_pw, 512, epre_pb, nullptr, bX, 512, Me, 512, 512, 0);

  // ===== encoder layers =====
  for (int i = 0; i < 8; ++i) {
    ln(bX, bH, e_ln1w + i * 512, e_ln1b + i * 512, Me);
    headlin(bH, e_qw + (long long)i * 262144, bQ, Me);
    headlin(bH, e_kw + (long long)i * 262144, bK, Me);
    headlin(bH, e_vw + (long long)i * 262144, bV, Me);
    attn(bQ, Te, bK, bV, Te, bO, 0);
    linear(bO, 512, e_pw + (long long)i * 262144, 512, e_pb + i * 512,
           bX, bX, 512, Me, 512, 512, 0);
    ln(bX, bH, e_ln2w + i * 512, e_ln2b + i * 512, Me);
    linear(bH, 512, e_f1w + (long long)i * 1048576, 2048, e_f1b + i * 2048,
           nullptr, bBIG, 2048, Me, 2048, 512, 1);
    linear(bBIG, 2048, e_f2w + (long long)i * 1048576, 512, e_f2b + i * 512,
           bX, bX, 512, Me, 512, 2048, 0);
  }
  // bX is now the encoder memory (B, 244, 512).

  // ===== decoder pre-net: d = relu(dec^T W1) relu(... W2) ... Wp =====
  {
    GemmP p = gdef();          // A[t][m] = dec_idx[b][m][t] via ka stride
    p.A = dec_idx; p.lda = 1; p.ka = 256;
    p.aOffB = 128LL * 256;
    p.B = dpre_w1; p.kSB = 512;
    p.bias = dpre_b1; p.relu = 1;
    p.C = bH; p.ldc = 512; p.cOffB = 256LL * 512;
    p.M = 256; p.N = 512; p.K = 128; p.Hdiv = 1;
    launch_gemm(p, 64, stream);
  }
  linear(bH, 512, dpre_w2, 512, dpre_b2, nullptr, bO, 512, Md, 512, 512, 1);
  linear(bO, 512, dpre_pw, 512, dpre_pb, nullptr, bD, 512, Md, 512, 512, 0);

  // ===== decoder layers =====
  for (int i = 0; i < 6; ++i) {
    ln(bD, bD, d_ln1w + i * 512, d_ln1b + i * 512, Md);   // reference: d = ln(d)
    headlin(bD, d_sqw + (long long)i * 262144, bQ, Md);
    headlin(bD, d_skw + (long long)i * 262144, bK, Md);
    headlin(bD, d_svw + (long long)i * 262144, bV, Md);
    attn(bQ, Td, bK, bV, Td, bO, 1);                      // causal self-attn
    linear(bO, 512, d_spw + (long long)i * 262144, 512, d_spb + i * 512,
           bD, bD, 512, Md, 512, 512, 0);
    ln(bD, bH, d_ln2w + i * 512, d_ln2b + i * 512, Md);
    headlin(bH, d_cqw + (long long)i * 262144, bQ, Md);
    headlin(bX, d_ckw + (long long)i * 262144, bK, Me);   // K,V from memory
    headlin(bX, d_cvw + (long long)i * 262144, bV, Me);
    attn(bQ, Td, bK, bV, Te, bO, 0);                      // cross-attn
    linear(bO, 512, d_cpw + (long long)i * 262144, 512, d_cpb + i * 512,
           bD, bD, 512, Md, 512, 512, 0);
    ln(bD, bH, d_ln3w + i * 512, d_ln3b + i * 512, Md);
    linear(bH, 512, d_f1w + (long long)i * 1048576, 2048, d_f1b + i * 2048,
           nullptr, bBIG, 2048, Md, 2048, 512, 1);
    linear(bBIG, 2048, d_f2w + (long long)i * 1048576, 512, d_f2b + i * 512,
           bD, bD, 512, Md, 512, 2048, 0);
  }

  // ===== head: last = d[:, -1, :]; out = (last W1 + b1) W2 + b2 =====
  linear(bD + 255LL * 512, 256LL * 512, m_w1, 512, m_b1, nullptr,
         bH, 512, 64, 512, 512, 0);
  linear(bH, 512, m_w2, 128, m_b2, nullptr,
         (float*)d_out, 128, 64, 128, 512, 0);
}
